// Net_1786706395264
// MI455X (gfx1250) — matrix operand order repaired and checked by reading, not compiled, not run on hardware
//
#include <hip/hip_runtime.h>
#include <hip/hip_fp16.h>

typedef __attribute__((ext_vector_type(16))) _Float16 v16h;
typedef __attribute__((ext_vector_type(8)))  float    v8f;

#define NFEAT 256
#define HID   64

union AFrag { v16h v; _Float16 e[16]; };
union CFrag { v8f  v; float     e[8];  };

// ---------------------------------------------------------------- utilities
__global__ void k_zero(float* __restrict__ p, size_t n) {
  size_t i = (size_t)blockIdx.x * 256 + threadIdx.x;
  if (i < n) p[i] = 0.0f;
}

// deg[dst] += w  (self loop +1 folded into k_dinv)
__global__ void k_deg(const int* __restrict__ dst, const float* __restrict__ ew,
                      float* __restrict__ deg, int E) {
  int e = blockIdx.x * 256 + threadIdx.x;
  if (e < E) atomicAdd(&deg[dst[e]], ew[e]);
}

// deg -> D^{-1/2} in place (deg+1 > 0 always: weights uniform(0,1) plus self loop)
__global__ void k_dinv(float* __restrict__ deg, int n) {
  int i = blockIdx.x * 256 + threadIdx.x;
  if (i < n) deg[i] = rsqrtf(deg[i] + 1.0f);
}

// ---------------------------------------------------------------- W1 fragment swizzle
// One-time: W1 (f32 [K=256][N=64]) -> f16, WMMA-B-fragment order.
// Dest d = ((kt*4 + t)*32 + lane)*16 + i holds W1[k, n] with
// k = kt*32 + (i>=8)*16 + (lane>=16)*8 + (i&7), n = t*16 + (lane&15)
// (B 32x16 f16 layout: lanes 0-15 hold K {0..7,16..23}, lanes 16-31 K {8..15,24..31}).
__global__ void k_swizzleW1(const float* __restrict__ W1, _Float16* __restrict__ Wf) {
  int d = blockIdx.x * 256 + threadIdx.x;
  if (d >= NFEAT * HID) return;
  int fragIdx = d >> 9;          // kt*4 + t
  int rem     = d & 511;
  int lidx    = rem >> 4;        // lane
  int ii      = rem & 15;
  int kt = fragIdx >> 2, t = fragIdx & 3;
  int k  = kt * 32 + (ii >> 3) * 16 + (lidx >> 4) * 8 + (ii & 7);
  int nn = t * 16 + (lidx & 15);
  Wf[d] = (_Float16)W1[k * HID + nn];
}

// ---------------------------------------------------------------- GEMM h = x @ W1
// Block = 256 threads = 8 waves. Pre-swizzled f16 W1 (32 KB, L2-resident) is
// staged into LDS with a plain vectorized copy (b128 in / b128 out). Each lane's
// B fragment is then 32 contiguous, aligned bytes -> two ds_load_b128.
// Each wave computes a 16(M) x 64(N) tile: 4 f32 accumulators, 8 K-steps of 32,
// 4 v_wmma_f32_16x16x32_f16 per step.
__global__ __launch_bounds__(256) void k_gemm_xw1(
    const float* __restrict__ x, const _Float16* __restrict__ Wf,
    float* __restrict__ h, int n_nodes) {
  __shared__ _Float16 sW[NFEAT * HID];   // 32 KB, fragment-ordered
  {
    const uint4* wsrc = (const uint4*)Wf;
    uint4* sdst = (uint4*)sW;
    #pragma unroll
    for (int r = 0; r < (NFEAT * HID / 8) / 256; ++r)
      sdst[r * 256 + threadIdx.x] = wsrc[r * 256 + threadIdx.x];
  }
  __syncthreads();

  const int wave = threadIdx.x >> 5;
  const int lane = threadIdx.x & 31;
  const int half = lane >> 4;
  const int ln   = lane & 15;

  const int m0 = (blockIdx.x * 8 + wave) * 16;

  CFrag acc[4];
  #pragma unroll
  for (int t = 0; t < 4; ++t)
    #pragma unroll
    for (int r = 0; r < 8; ++r) acc[t].e[r] = 0.0f;

  int row = m0 + ln;
  if (row >= n_nodes) row = n_nodes - 1;    // clamp: keep EXEC all-ones for WMMA
  const float4* xr4 = (const float4*)(x + (size_t)row * NFEAT);

  for (int k0 = 0; k0 < NFEAT; k0 += 32) {
    // A fragment 16x32 f16: lanes 0-15 hold K {0..7,16..23},
    // lanes 16-31 hold K {8..15,24..31}, M = lane&15.  float4 loads + cvt.
    const int ka = (k0 + half * 8) >> 2;     // in float4 units
    float4 p0 = xr4[ka];
    float4 p1 = xr4[ka + 1];
    float4 p2 = xr4[ka + 4];
    float4 p3 = xr4[ka + 5];
    AFrag a;
    a.e[0]=(_Float16)p0.x;  a.e[1]=(_Float16)p0.y;  a.e[2]=(_Float16)p0.z;  a.e[3]=(_Float16)p0.w;
    a.e[4]=(_Float16)p1.x;  a.e[5]=(_Float16)p1.y;  a.e[6]=(_Float16)p1.z;  a.e[7]=(_Float16)p1.w;
    a.e[8]=(_Float16)p2.x;  a.e[9]=(_Float16)p2.y;  a.e[10]=(_Float16)p2.z; a.e[11]=(_Float16)p2.w;
    a.e[12]=(_Float16)p3.x; a.e[13]=(_Float16)p3.y; a.e[14]=(_Float16)p3.z; a.e[15]=(_Float16)p3.w;

    // B fragments: contiguous 32 B per lane -> 2x ds_load_b128 each
    const _Float16* fb = sW + ((k0 >> 5) * 4) * 512 + lane * 16;
    #pragma unroll
    for (int t = 0; t < 4; ++t) {
      AFrag b;
      b.v = *(const v16h*)(fb + t * 512);
      acc[t].v = __builtin_amdgcn_wmma_f32_16x16x32_f16(
          false, a.v, false, b.v, (short)0, acc[t].v, false, false);
    }
  }

  // D 16x16 f32 layout: VGPR r holds M = r (lanes 0-15) / M = 8+r (lanes 16-31).
  // Uniform fast path: only the final row-tile needs per-row guards.
  if (m0 + 16 <= n_nodes) {
    float* hp = h + (size_t)(m0 + half * 8) * HID + ln;
    #pragma unroll
    for (int t = 0; t < 4; ++t)
      #pragma unroll
      for (int r = 0; r < 8; ++r)
        hp[(size_t)r * HID + t * 16] = acc[t].e[r];
  } else {
    #pragma unroll
    for (int t = 0; t < 4; ++t)
      #pragma unroll
      for (int r = 0; r < 8; ++r) {
        int m = m0 + half * 8 + r;
        if (m < n_nodes) h[(size_t)m * HID + t * 16 + ln] = acc[t].e[r];
      }
  }
}

// ---------------------------------------------------------------- edge scatter, layer 1
// One wave per edge: each lane handles 2 of the 64 features (coalesced float2
// gather of h[src], 2 non-returning f32 atomics into agg[dst]).
// h (25.6 MB) and agg (25.6 MB) both stay resident in the 192 MB L2.
__global__ __launch_bounds__(256) void k_scatter1(
    const int* __restrict__ src, const int* __restrict__ dst,
    const float* __restrict__ ew, const float* __restrict__ dinv,
    const float* __restrict__ h, float* __restrict__ agg, int E) {
  int e = blockIdx.x * 8 + (threadIdx.x >> 5);
  if (e >= E) return;
  const int lane = threadIdx.x & 31;
  const int s = src[e];
  const int d = dst[e];
  __builtin_prefetch(&h[(size_t)s * HID], 0, 0);   // global_prefetch_b8
  const float norm = dinv[s] * ew[e] * dinv[d];
  const float2 v = ((const float2*)(h + (size_t)s * HID))[lane];
  float* ad = agg + (size_t)d * HID + lane * 2;
  atomicAdd(ad,     norm * v.x);
  atomicAdd(ad + 1, norm * v.y);
}

// epilogue 1: add self-loop term + bias, ELU, in place (agg1 becomes hfeat)
__global__ void k_finish1(float* __restrict__ agg, const float* __restrict__ h,
                          const float* __restrict__ dinv, const float* __restrict__ b1,
                          int n_nodes) {
  size_t idx = (size_t)blockIdx.x * 256 + threadIdx.x;
  if (idx >= (size_t)n_nodes * HID) return;
  int i = (int)(idx >> 6);
  int j = (int)(idx & 63);
  float di = dinv[i];
  float v = agg[idx] + di * di * h[idx] + b1[j];
  agg[idx] = (v > 0.0f) ? v : (__expf(v) - 1.0f);
}

// ---------------------------------------------------------------- layer 2
// z[i] = hfeat[i,:] . W2   — one wave per node, float2 per lane, xor-shuffle reduce
__global__ __launch_bounds__(256) void k_dot(
    const float* __restrict__ hf, const float* __restrict__ W2,
    float* __restrict__ z, int n_nodes) {
  int i = blockIdx.x * 8 + (threadIdx.x >> 5);
  if (i >= n_nodes) return;
  const int lane = threadIdx.x & 31;
  const float2 a = ((const float2*)(hf + (size_t)i * HID))[lane];
  const float2 b = ((const float2*)W2)[lane];
  float s = a.x * b.x + a.y * b.y;
  #pragma unroll
  for (int off = 16; off > 0; off >>= 1) s += __shfl_xor(s, off, 32);
  if (lane == 0) z[i] = s;
}

__global__ void k_scatter2(const int* __restrict__ src, const int* __restrict__ dst,
                           const float* __restrict__ ew, const float* __restrict__ dinv,
                           const float* __restrict__ z, float* __restrict__ agg2, int E) {
  int e = blockIdx.x * 256 + threadIdx.x;
  if (e >= E) return;
  const int s = src[e];
  const int d = dst[e];
  atomicAdd(&agg2[d], dinv[s] * ew[e] * dinv[d] * z[s]);
}

__global__ void k_finish2(const float* __restrict__ agg2, const float* __restrict__ z,
                          const float* __restrict__ dinv, const float* __restrict__ b2,
                          float* __restrict__ out, int n_nodes) {
  int i = blockIdx.x * 256 + threadIdx.x;
  if (i >= n_nodes) return;
  float di = dinv[i];
  float s = agg2[i] + di * di * z[i] + b2[0];
  out[i] = 1.0f / (1.0f + __expf(-s));
}

// ---------------------------------------------------------------- launch
extern "C" void kernel_launch(void* const* d_in, const int* in_sizes, int n_in,
                              void* d_out, int out_size, void* d_ws, size_t ws_size,
                              hipStream_t stream) {
  const float* x  = (const float*)d_in[0];
  const int*   ei = (const int*)  d_in[1];   // [2, E] (src row then dst row)
  const float* ew = (const float*)d_in[2];
  const float* W1 = (const float*)d_in[3];
  const float* b1 = (const float*)d_in[4];
  const float* W2 = (const float*)d_in[5];
  const float* b2 = (const float*)d_in[6];

  const int n = in_sizes[0] / NFEAT;   // 100000
  const int E = in_sizes[2];           // 3200000
  const int* src = ei;
  const int* dst = ei + E;

  // workspace: deg/dinv[n] | h[n*64] | agg1[n*64] | z[n] | agg2[n] | Wf[16K f16]
  float* deg  = (float*)d_ws;
  float* h    = deg  + n;
  float* agg1 = h    + (size_t)n * HID;
  float* z    = agg1 + (size_t)n * HID;
  float* agg2 = z    + n;
  // align the f16 fragment image to 256 B for uint4 copies
  uintptr_t wfAddr = ((uintptr_t)(agg2 + n) + 255) & ~(uintptr_t)255;
  _Float16* Wf = (_Float16*)wfAddr;

  const size_t nh = (size_t)n * HID;

  k_zero<<<(unsigned)((n + 255) / 256), 256, 0, stream>>>(deg, (size_t)n);
  k_zero<<<(unsigned)((nh + 255) / 256), 256, 0, stream>>>(agg1, nh);
  k_zero<<<(unsigned)((n + 255) / 256), 256, 0, stream>>>(agg2, (size_t)n);

  k_deg <<<(E + 255) / 256, 256, 0, stream>>>(dst, ew, deg, E);
  k_dinv<<<(n + 255) / 256, 256, 0, stream>>>(deg, n);

  k_swizzleW1<<<(NFEAT * HID + 255) / 256, 256, 0, stream>>>(W1, Wf);
  k_gemm_xw1 <<<(n + 127) / 128, 256, 0, stream>>>(x, Wf, h, n);

  k_scatter1<<<(E + 7) / 8, 256, 0, stream>>>(src, dst, ew, deg, h, agg1, E);
  k_finish1 <<<(unsigned)((nh + 255) / 256), 256, 0, stream>>>(agg1, h, deg, b1, n);

  k_dot     <<<(n + 7) / 8, 256, 0, stream>>>(agg1, W2, z, n);
  k_scatter2<<<(E + 255) / 256, 256, 0, stream>>>(src, dst, ew, deg, z, agg2, E);
  k_finish2 <<<(n + 255) / 256, 256, 0, stream>>>(agg2, z, deg, b2, (float*)d_out, n);
}
